// Raindrop_v2_26388279066958
// MI455X (gfx1250) — compile-verified, matrix operand order checked by reading, not run on hardware
//
#include <hip/hip_runtime.h>
#include <hip/hip_bf16.h>
#include <math.h>

typedef __attribute__((ext_vector_type(16))) _Float16 v16h;
typedef __attribute__((ext_vector_type(8)))  _Float16 v8h;
typedef __attribute__((ext_vector_type(8)))  float    v8f;

#define BB   8
#define LL   215
#define NINP 36
#define DPE  16
#define DTR  160
#define NHID_ 256
#define EE   1296      // 36*36 edges
#define KK2  648       // kept edges
#define CC   860       // L * d_ob
#define CPAD 864       // padded to mult of 32
#define RW   6880      // C*8 rows of W_inc
#define ROWS 1720      // B*L
#define RPAD 1728      // padded rows
#define TPAD 224       // L padded to mult of 16/32
#define DHP  64        // head dim 40 padded to 64
#define NH   4

static inline int cdiv(int a, int b){ return (a + b - 1) / b; }

// ---------------------------------------------------------------------------
// WMMA fragment load: 16-bit A/B 16x32 layout (ISA 7.12.2):
// lane&15 selects row(A)/col(B); lane>>4 selects K-half {0..7,16..23} vs {8..15,24..31}
// caller passes p = base + (lane&15)*ld + (lane>>4)*8 + kchunk ; halves at p[0..7], p[16..23]
// ---------------------------------------------------------------------------
__device__ inline v16h ldfrag(const _Float16* p) {
  v8h lo = *reinterpret_cast<const v8h*>(p);
  v8h hi = *reinterpret_cast<const v8h*>(p + 16);
  v16h r;
#pragma unroll
  for (int i = 0; i < 8; ++i) { r[i] = lo[i]; r[i + 8] = hi[i]; }
  return r;
}

// Generic f16->f32 WMMA GEMM:  D[m,n] = sum_k A[m,k] * Bw[n,k]  (+bias[n]) (*rowscale[m])
// grid = (Mtiles, Ntiles, batch), block = 32 (one wave per 16x16 output tile).
__global__ void k_gemm_wmma(const _Float16* __restrict__ A,  int lda,  unsigned long long sA,
                            const _Float16* __restrict__ Bw, int ldb,  unsigned long long sB,
                            const float*    __restrict__ bias,
                            const float*    __restrict__ rowscale,
                            float*          __restrict__ D,  int ldd,  unsigned long long sD,
                            _Float16*       __restrict__ Dh, int lddh, unsigned long long sDh,
                            int Mvalid, int Nvalid, int K, int relu)
{
  int z = blockIdx.z;
  A  += (size_t)z * sA;
  Bw += (size_t)z * sB;
  if (D)  D  += (size_t)z * sD;
  if (Dh) Dh += (size_t)z * sDh;

  int lane  = threadIdx.x;
  int mBase = blockIdx.x * 16, nBase = blockIdx.y * 16;
  int r15   = lane & 15;
  int khalf = (lane >> 4) * 8;
  const _Float16* arow = A  + (size_t)(mBase + r15) * lda + khalf;
  const _Float16* brow = Bw + (size_t)(nBase + r15) * ldb + khalf;

  v8f c = {};
  for (int k = 0; k < K; k += 32) {
    if (k + 64 <= K) { __builtin_prefetch(arow + k + 32, 0, 1); __builtin_prefetch(brow + k + 32, 0, 1); }
    v16h a = ldfrag(arow + k);
    v16h b = ldfrag(brow + k);
    c = __builtin_amdgcn_wmma_f32_16x16x32_f16(false, a, false, b, (short)0, c, false, false);
  }

  // C/D layout: col = nBase + (lane&15); rows mBase + (lane>=16 ? 8 : 0) + r
  int col  = nBase + (lane & 15);
  int row0 = mBase + (lane >> 4) * 8;
  if (col < Nvalid) {
    float bv = bias ? bias[col] : 0.f;
#pragma unroll
    for (int r = 0; r < 8; ++r) {
      int row = row0 + r;
      if (row < Mvalid) {
        float v = c[r] + bv;
        if (relu) v = fmaxf(v, 0.f);
        if (rowscale) v *= rowscale[row];
        if (D)  D [(size_t)row * ldd  + col] = v;
        if (Dh) Dh[(size_t)row * lddh + col] = (_Float16)v;
      }
    }
  }
}

// ---------------------------------------------------------------------------
// small utility kernels
// ---------------------------------------------------------------------------
__global__ void k_zero_f32(float* p, int n){ int i = blockIdx.x*blockDim.x+threadIdx.x; if (i < n) p[i] = 0.f; }
__global__ void k_zero_f16(_Float16* p, int n){ int i = blockIdx.x*blockDim.x+threadIdx.x; if (i < n) p[i] = (_Float16)0.f; }

// f32[R,Cs] -> f16[Rd,Cd] zero-padded
__global__ void k_cvt_pad(const float* __restrict__ src, _Float16* __restrict__ dst,
                          int R, int Cs, int Rd, int Cd)
{
  int i = blockIdx.x*blockDim.x + threadIdx.x;
  if (i >= Rd * Cd) return;
  int r = i / Cd, c = i % Cd;
  float v = (r < R && c < Cs) ? src[(size_t)r * Cs + c] : 0.f;
  dst[i] = (_Float16)v;
}

__global__ void k_lengths(const int* __restrict__ et, int* __restrict__ lengths)
{
  __shared__ int red[256];
  int b = blockIdx.x, tid = threadIdx.x;
  int c = 0;
  for (int t = tid; t < LL; t += 256) c += (et[b*LL + t] != 0);
  red[tid] = c; __syncthreads();
  for (int s = 128; s > 0; s >>= 1) { if (tid < s) red[tid] += red[tid + s]; __syncthreads(); }
  if (tid == 0) lengths[b] = red[0];
}

__global__ void k_pe_nonpad(const int* __restrict__ et, const float* __restrict__ etime,
                            float* __restrict__ pe, float* __restrict__ nonpad)
{
  int i = blockIdx.x*blockDim.x + threadIdx.x;
  if (i >= BB * LL) return;
  float tm = etime[i];
#pragma unroll
  for (int s = 0; s < 8; ++s) {
    float ts = powf(215.f, (float)s / 7.f);
    float sc = tm / ts;
    pe[(size_t)i * 16 + s]     = sinf(sc);
    pe[(size_t)i * 16 + 8 + s] = cosf(sc);
  }
  nonpad[i] = (et[i] != 0) ? 1.f : 0.f;
}

// x[b][n][t*4+o] = relu(R_u[n*4+o]) where n = max(et-1,0); both f32 and padded f16 copies
__global__ void k_scatter_x(const int* __restrict__ et, const float* __restrict__ Ru,
                            float* __restrict__ xbuf, _Float16* __restrict__ xh)
{
  int i = blockIdx.x*blockDim.x + threadIdx.x;
  if (i >= BB * LL) return;
  int b = i / LL, t = i % LL;
  int idx = et[i] - 1; if (idx < 0) idx = 0;
#pragma unroll
  for (int o = 0; o < 4; ++o) {
    float v = fmaxf(Ru[idx*4 + o], 0.f);
    xbuf[((size_t)b*NINP + idx)*CPAD + t*4 + o] = v;
    xh  [((size_t)b*48   + idx)*CPAD + t*4 + o] = (_Float16)v;
  }
}

// beta_node[b][n][t] = (sum_{s<16} hW[b][n][t*32+s]*map_w[n][s] + hW[...t*32+16+s]*pe[b][t][s]) / 32
__global__ void k_beta(const float* __restrict__ hW, const float* __restrict__ map_w,
                       const float* __restrict__ pe, float* __restrict__ beta_node)
{
  int i = blockIdx.x*blockDim.x + threadIdx.x;
  if (i >= BB * NINP * LL) return;
  int t = i % LL; int bn = i / LL; int n = bn % NINP; int b = bn / NINP;
  const float* hw = hW + ((size_t)b*48 + n) * RW + t*32;
  const float* mw = map_w + n*16;
  const float* p  = pe + ((size_t)b*LL + t) * 16;
  float acc = 0.f;
#pragma unroll
  for (int s = 0; s < 16; ++s) acc += hw[s]*mw[s] + hw[16+s]*p[s];
  beta_node[((size_t)b*NINP + n)*LL + t] = acc * (1.f/32.f);
}

__global__ void k_betamean(const float* __restrict__ beta_node, float* __restrict__ bm)
{
  __shared__ float red[256];
  int blk = blockIdx.x;   // b*36+n
  int tid = threadIdx.x;
  float s = 0.f;
  for (int t = tid; t < LL; t += 256) s += beta_node[(size_t)blk*LL + t];
  red[tid] = s; __syncthreads();
  for (int st = 128; st > 0; st >>= 1) { if (tid < st) red[tid] += red[tid + st]; __syncthreads(); }
  if (tid == 0) bm[blk] = red[0] / (float)LL;
}

// gm[b][e] = ew[e] * betamean[b][e%36] ; ew from global_structure with diag=1
__global__ void k_gm(const float* __restrict__ gs, const float* __restrict__ bm, float* __restrict__ gm)
{
  int i = blockIdx.x*blockDim.x + threadIdx.x;
  if (i >= BB * EE) return;
  int e = i % EE; int b = i / EE;
  int sr = e / NINP, ds = e % NINP;
  float ew = (sr == ds) ? 1.f : gs[e];
  gm[i] = ew * bm[b*NINP + ds];
}

// stable descending argsort rank; keep rank<K, scatter sorted values (== alpha) to Asort
__global__ void k_rank(const float* __restrict__ gm, int* __restrict__ kept, float* __restrict__ Asort)
{
  int i = blockIdx.x*blockDim.x + threadIdx.x;
  if (i >= BB * EE) return;
  int e = i % EE; int b = i / EE;
  const float* g = gm + (size_t)b * EE;
  float v = g[e];
  int cnt = 0;
  for (int e2 = 0; e2 < EE; ++e2) {
    float u = g[e2];
    cnt += (u > v) || (u == v && e2 < e);
  }
  if (cnt < KK2) { kept[i] = 1; Asort[b*KK2 + cnt] = v; }
  else kept[i] = 0;
}

// h1[b][n][c] = sum over kept edges e=m*36+n of relu(x[b][m][c]) * relu(ew[e]*beta_node[b][n][c>>2])
__global__ void k_h1(const float* __restrict__ xbuf, const float* __restrict__ beta_node,
                     const float* __restrict__ gs, const int* __restrict__ kept,
                     float* __restrict__ h1)
{
  int blk = blockIdx.x;  // b*36+n
  int n = blk % NINP, b = blk / NINP;
  for (int c = threadIdx.x; c < CC; c += blockDim.x) {
    float bt = beta_node[((size_t)b*NINP + n)*LL + (c >> 2)];
    float acc = 0.f;
    for (int m = 0; m < NINP; ++m) {
      int e = m*NINP + n;
      if (kept[b*EE + e]) {
        float ew = (m == n) ? 1.f : gs[e];
        float g  = fmaxf(ew * bt, 0.f);
        acc += fmaxf(xbuf[((size_t)b*NINP + m)*CPAD + c], 0.f) * g;
      }
    }
    h1[((size_t)b*NINP + n)*CPAD + c] = acc;
  }
}

// h2[b][n][c] = sum over kept edges e=m*36+n of relu(h1[b][m][c]) * relu(alpha[e]);  alpha == gm
__global__ void k_h2(const float* __restrict__ h1, const float* __restrict__ gm,
                     const int* __restrict__ kept, float* __restrict__ h2)
{
  int blk = blockIdx.x;
  int n = blk % NINP, b = blk / NINP;
  for (int c = threadIdx.x; c < CC; c += blockDim.x) {
    float acc = 0.f;
    for (int m = 0; m < NINP; ++m) {
      int e = m*NINP + n;
      if (kept[b*EE + e]) {
        float a = fmaxf(gm[b*EE + e], 0.f);
        acc += fmaxf(h1[((size_t)b*NINP + m)*CPAD + c], 0.f) * a;
      }
    }
    h2[((size_t)b*NINP + n)*CPAD + c] = acc;
  }
}

// x0[b*L+t][j] : j<144 -> h2[b][j>>2][t*4+(j&3)] ; else pe[b][t][j-144]
__global__ void k_build_x0(const float* __restrict__ h2, const float* __restrict__ pe,
                           float* __restrict__ x0, _Float16* __restrict__ x0h)
{
  int i = blockIdx.x*blockDim.x + threadIdx.x;
  if (i >= ROWS * DTR) return;
  int row = i / DTR, j = i % DTR;
  int b = row / LL, t = row % LL;
  float v;
  if (j < 144) v = h2[((size_t)b*NINP + (j >> 2))*CPAD + t*4 + (j & 3)];
  else         v = pe[(size_t)row * 16 + (j - 144)];
  x0[i]  = v;
  x0h[i] = (_Float16)v;
}

// pack q,k into [bh][224][64], v transposed into [bh][64][224] (zero padded)
__global__ void k_pack_qkv(const _Float16* __restrict__ qkvh,
                           _Float16* __restrict__ qp, _Float16* __restrict__ kp,
                           _Float16* __restrict__ vtp)
{
  int i = blockIdx.x*blockDim.x + threadIdx.x;
  if (i >= BB*NH*TPAD*DHP) return;
  int d = i % DHP; int t = (i / DHP) % TPAD; int bh = i / (DHP*TPAD);
  int b = bh >> 2, h = bh & 3;
  float q = 0.f, k = 0.f, v = 0.f;
  if (t < LL && d < 40) {
    size_t r = (size_t)(b*LL + t)*480 + h*40 + d;
    q = (float)qkvh[r]; k = (float)qkvh[r + 160]; v = (float)qkvh[r + 320];
  }
  qp [((size_t)bh*TPAD + t)*DHP + d] = (_Float16)q;
  kp [((size_t)bh*TPAD + t)*DHP + d] = (_Float16)k;
  vtp[((size_t)bh*DHP  + d)*TPAD + t] = (_Float16)v;
}

// scaled masked softmax over one score row -> f16 probabilities
__global__ void k_softmax(const float* __restrict__ scores, const int* __restrict__ lengths,
                          _Float16* __restrict__ pp)
{
  __shared__ float red[256];
  __shared__ float smax, ssum;
  int row = blockIdx.x;                 // (bh)*224 + t
  int b = row / (NH * TPAD);
  int tid = threadIdx.x;
  const float* s = scores + (size_t)row * TPAD;
  int len = lengths[b];
  const float scale = 0.158113883008419f;  // 1/sqrt(40)
  float x = -3.0e38f;
  if (tid < TPAD) { x = s[tid] * scale; if (tid >= len) x = -1.0e9f; }
  red[tid] = x; __syncthreads();
  for (int st = 128; st > 0; st >>= 1) { if (tid < st) red[tid] = fmaxf(red[tid], red[tid + st]); __syncthreads(); }
  if (tid == 0) smax = red[0]; __syncthreads();
  float e = (tid < TPAD) ? expf(x - smax) : 0.f;
  red[tid] = e; __syncthreads();
  for (int st = 128; st > 0; st >>= 1) { if (tid < st) red[tid] += red[tid + st]; __syncthreads(); }
  if (tid == 0) ssum = red[0]; __syncthreads();
  if (tid < TPAD) pp[(size_t)row * TPAD + tid] = (_Float16)(e / ssum);
}

// merge per-head ctx tiles [bh][224][64] -> [row][160] f16
__global__ void k_ctx_gather(const _Float16* __restrict__ ctxp, _Float16* __restrict__ ctxh)
{
  int i = blockIdx.x*blockDim.x + threadIdx.x;
  if (i >= ROWS * DTR) return;
  int row = i / DTR, j = i % DTR;
  int h = j / 40, d = j % 40;
  int b = row / LL, t = row % LL;
  ctxh[i] = ctxp[(((size_t)(b*NH + h))*TPAD + t)*DHP + d];
}

// xout = LayerNorm(xin + delta) with (w,b); writes f32 and f16 mirror. one block per row.
__global__ void k_residual_ln(const float* __restrict__ xin, const float* __restrict__ delta,
                              const float* __restrict__ w, const float* __restrict__ bparm,
                              float* __restrict__ xout, _Float16* __restrict__ xouth)
{
  __shared__ float red[256];
  __shared__ float s0, s1;
  int row = blockIdx.x, tid = threadIdx.x;
  float val = 0.f;
  if (tid < DTR) val = xin[(size_t)row*DTR + tid] + delta[(size_t)row*DTR + tid];
  red[tid] = (tid < DTR) ? val : 0.f; __syncthreads();
  for (int s = 128; s > 0; s >>= 1) { if (tid < s) red[tid] += red[tid + s]; __syncthreads(); }
  if (tid == 0) s0 = red[0] / (float)DTR; __syncthreads();
  float d = (tid < DTR) ? (val - s0) : 0.f;
  red[tid] = d * d; __syncthreads();
  for (int s = 128; s > 0; s >>= 1) { if (tid < s) red[tid] += red[tid + s]; __syncthreads(); }
  if (tid == 0) s1 = rsqrtf(red[0] / (float)DTR + 1e-5f); __syncthreads();
  if (tid < DTR) {
    float o = d * s1 * w[tid] + bparm[tid];
    xout [(size_t)row*DTR + tid] = o;
    xouth[(size_t)row*DTR + tid] = (_Float16)o;
  }
}

// distance = mean over BxB pairs of ||A[i]-A[j]|| (0 on diagonal / zero distance)
__global__ void k_distance(const float* __restrict__ Asort, float* __restrict__ out)
{
  __shared__ float red[256];
  __shared__ float acc;
  int tid = threadIdx.x;
  if (tid == 0) acc = 0.f;
  __syncthreads();
  for (int p = 0; p < BB*BB; ++p) {
    int i = p / BB, j = p % BB;
    float s = 0.f;
    for (int k = tid; k < KK2; k += 256) {
      float d = Asort[i*KK2 + k] - Asort[j*KK2 + k];
      s += d * d;
    }
    red[tid] = s; __syncthreads();
    for (int st = 128; st > 0; st >>= 1) { if (tid < st) red[tid] += red[tid + st]; __syncthreads(); }
    if (tid == 0 && red[0] > 0.f) acc += sqrtf(red[0]);
    __syncthreads();
  }
  if (tid == 0) out[0] = acc / (float)(BB*BB);
}

// ---------------------------------------------------------------------------
extern "C" void kernel_launch(void* const* d_in, const int* in_sizes, int n_in,
                              void* d_out, int out_size, void* d_ws, size_t ws_size,
                              hipStream_t stream)
{
  (void)in_sizes; (void)n_in; (void)out_size; (void)ws_size;

  const int*   event_type = (const int*)  d_in[0];
  const float* event_time = (const float*)d_in[1];
  const float* R_u        = (const float*)d_in[2];
  const float* gs         = (const float*)d_in[3];
  const float* W_inc      = (const float*)d_in[4];
  const float* b_inc      = (const float*)d_in[5];
  const float* map_w      = (const float*)d_in[6];
  const float* w_time     = (const float*)d_in[7];
  const float* w_type     = (const float*)d_in[8];
  const float* Lw[2][12];
  for (int l = 0; l < 2; ++l)
    for (int j = 0; j < 12; ++j)
      Lw[l][j] = (const float*)d_in[9 + l*12 + j];

  float* outF      = (float*)d_out;
  float* outMain   = outF;                       // [8,215,160]
  float* outType   = outF + ROWS*DTR;            // [8,215,36]
  float* outTime   = outType + ROWS*NINP;        // [8,215,1]
  float* outDist   = outTime + ROWS;             // [1]

  // ---- workspace bump allocator (256B aligned) ----
  size_t off = 0;
  auto alloc = [&](size_t bytes) -> void* {
    void* p = (char*)d_ws + off;
    off += (bytes + 255) & ~(size_t)255;
    return p;
  };
  _Float16* winch   = (_Float16*)alloc((size_t)RW * CPAD * 2);
  _Float16* qkvwh[2]; _Float16* outwh[2]; _Float16* ff1wh[2]; _Float16* ff2wh[2];
  for (int l = 0; l < 2; ++l) {
    qkvwh[l] = (_Float16*)alloc((size_t)480 * DTR * 2);
    outwh[l] = (_Float16*)alloc((size_t)DTR * DTR * 2);
    ff1wh[l] = (_Float16*)alloc((size_t)NHID_ * DTR * 2);
    ff2wh[l] = (_Float16*)alloc((size_t)DTR * NHID_ * 2);
  }
  _Float16* wtypeh  = (_Float16*)alloc((size_t)48 * DTR * 2);
  _Float16* wtimeh  = (_Float16*)alloc((size_t)16 * DTR * 2);
  float*    xbuf    = (float*)   alloc((size_t)BB * NINP * CPAD * 4);
  _Float16* xh      = (_Float16*)alloc((size_t)BB * 48 * CPAD * 2);
  float*    hW      = (float*)   alloc((size_t)BB * 48 * RW * 4);
  float*    pe      = (float*)   alloc((size_t)ROWS * DPE * 4);
  float*    nonpad  = (float*)   alloc((size_t)ROWS * 4);
  int*      lengths = (int*)     alloc((size_t)BB * 4);
  float*    beta_n  = (float*)   alloc((size_t)BB * NINP * LL * 4);
  float*    bm      = (float*)   alloc((size_t)BB * NINP * 4);
  float*    gm      = (float*)   alloc((size_t)BB * EE * 4);
  int*      kept    = (int*)     alloc((size_t)BB * EE * 4);
  float*    Asort   = (float*)   alloc((size_t)BB * KK2 * 4);
  float*    h1      = (float*)   alloc((size_t)BB * NINP * CPAD * 4);
  float*    h2      = (float*)   alloc((size_t)BB * NINP * CPAD * 4);
  float*    bufA    = (float*)   alloc((size_t)RPAD * DTR * 4);
  _Float16* bufAh   = (_Float16*)alloc((size_t)RPAD * DTR * 2);
  float*    bufB    = (float*)   alloc((size_t)RPAD * DTR * 4);
  _Float16* bufBh   = (_Float16*)alloc((size_t)RPAD * DTR * 2);
  float*    proj    = (float*)   alloc((size_t)RPAD * DTR * 4);
  _Float16* qkvh    = (_Float16*)alloc((size_t)RPAD * 480 * 2);
  _Float16* qp      = (_Float16*)alloc((size_t)BB*NH*TPAD*DHP * 2);
  _Float16* kp      = (_Float16*)alloc((size_t)BB*NH*TPAD*DHP * 2);
  _Float16* vtp     = (_Float16*)alloc((size_t)BB*NH*DHP*TPAD * 2);
  float*    scores  = (float*)   alloc((size_t)BB*NH*TPAD*TPAD * 4);
  _Float16* pp      = (_Float16*)alloc((size_t)BB*NH*TPAD*TPAD * 2);
  _Float16* ctxp    = (_Float16*)alloc((size_t)BB*NH*TPAD*DHP * 2);
  _Float16* ctxh    = (_Float16*)alloc((size_t)RPAD * DTR * 2);
  _Float16* ffh     = (_Float16*)alloc((size_t)RPAD * NHID_ * 2);

  auto gemm = [&](const _Float16* A, int lda, size_t sA,
                  const _Float16* Bw, int ldb, size_t sB,
                  const float* bias, const float* rowscale,
                  float* D, int ldd, size_t sD,
                  _Float16* Dh, int lddh, size_t sDh,
                  int Mt, int Nt, int Mv, int Nv, int K, int relu, int batch) {
    dim3 g(Mt, Nt, batch);
    k_gemm_wmma<<<g, dim3(32), 0, stream>>>(A, lda, sA, Bw, ldb, sB, bias, rowscale,
                                            D, ldd, sD, Dh, lddh, sDh, Mv, Nv, K, relu);
  };

  // ---- weight conversions to f16 (with zero padding) ----
  k_cvt_pad<<<cdiv(RW*CPAD,256),256,0,stream>>>(W_inc, winch, RW, CC, RW, CPAD);
  for (int l = 0; l < 2; ++l) {
    k_cvt_pad<<<cdiv(480*DTR,256),256,0,stream>>>(Lw[l][0], qkvwh[l], 480, DTR, 480, DTR);
    k_cvt_pad<<<cdiv(DTR*DTR,256),256,0,stream>>>(Lw[l][2], outwh[l], DTR, DTR, DTR, DTR);
    k_cvt_pad<<<cdiv(NHID_*DTR,256),256,0,stream>>>(Lw[l][4], ff1wh[l], NHID_, DTR, NHID_, DTR);
    k_cvt_pad<<<cdiv(DTR*NHID_,256),256,0,stream>>>(Lw[l][6], ff2wh[l], DTR, NHID_, DTR, NHID_);
  }
  k_cvt_pad<<<cdiv(48*DTR,256),256,0,stream>>>(w_type, wtypeh, NINP, DTR, 48, DTR);
  k_cvt_pad<<<cdiv(16*DTR,256),256,0,stream>>>(w_time, wtimeh, 1, DTR, 16, DTR);

  // ---- input prep ----
  k_zero_f32<<<cdiv(BB*NINP*CPAD,256),256,0,stream>>>(xbuf, BB*NINP*CPAD);
  k_zero_f16<<<cdiv(BB*48*CPAD,256),256,0,stream>>>(xh, BB*48*CPAD);
  k_zero_f16<<<cdiv(RPAD*DTR,256),256,0,stream>>>(bufAh, RPAD*DTR);
  k_lengths<<<BB,256,0,stream>>>(event_type, lengths);
  k_pe_nonpad<<<cdiv(BB*LL,256),256,0,stream>>>(event_type, event_time, pe, nonpad);
  k_scatter_x<<<cdiv(BB*LL,256),256,0,stream>>>(event_type, R_u, xbuf, xh);

  // ---- graph propagation ----
  // hW[b] = x[b] (36x864) @ W_inc^T (864x6880) + b_inc   (WMMA, batch=8)
  gemm(xh, CPAD, (size_t)48*CPAD, winch, CPAD, 0, b_inc, nullptr,
       hW, RW, (size_t)48*RW, nullptr, 0, 0, 3, RW/16, 48, RW, CPAD, 0, BB);
  k_beta<<<cdiv(BB*NINP*LL,256),256,0,stream>>>(hW, map_w, pe, beta_n);
  k_betamean<<<BB*NINP,256,0,stream>>>(beta_n, bm);
  k_gm<<<cdiv(BB*EE,256),256,0,stream>>>(gs, bm, gm);
  k_rank<<<cdiv(BB*EE,256),256,0,stream>>>(gm, kept, Asort);
  k_h1<<<BB*NINP,256,0,stream>>>(xbuf, beta_n, gs, kept, h1);
  k_h2<<<BB*NINP,256,0,stream>>>(h1, gm, kept, h2);
  k_build_x0<<<cdiv(ROWS*DTR,256),256,0,stream>>>(h2, pe, bufA, bufAh);
  k_distance<<<1,256,0,stream>>>(Asort, outDist);

  // ---- transformer encoder layers ----
  for (int l = 0; l < 2; ++l) {
    // QKV: [1720,160] x [480,160]^T -> f16 [1720,480]
    gemm(bufAh, DTR, 0, qkvwh[l], DTR, 0, Lw[l][1], nullptr,
         nullptr, 0, 0, qkvh, 480, 0, RPAD/16, 480/16, ROWS, 480, DTR, 0, 1);
    k_pack_qkv<<<cdiv(BB*NH*TPAD*DHP,256),256,0,stream>>>(qkvh, qp, kp, vtp);
    // scores = q @ k^T  (batch = 32 heads)
    gemm(qp, DHP, (size_t)TPAD*DHP, kp, DHP, (size_t)TPAD*DHP, nullptr, nullptr,
         scores, TPAD, (size_t)TPAD*TPAD, nullptr, 0, 0,
         TPAD/16, TPAD/16, TPAD, TPAD, DHP, 0, BB*NH);
    k_softmax<<<BB*NH*TPAD,256,0,stream>>>(scores, lengths, pp);
    // ctx = P @ V  (K = 224)
    gemm(pp, TPAD, (size_t)TPAD*TPAD, vtp, TPAD, (size_t)DHP*TPAD, nullptr, nullptr,
         nullptr, 0, 0, ctxp, DHP, (size_t)TPAD*DHP,
         TPAD/16, DHP/16, TPAD, DHP, TPAD, 0, BB*NH);
    k_ctx_gather<<<cdiv(ROWS*DTR,256),256,0,stream>>>(ctxp, ctxh);
    // out-proj
    gemm(ctxh, DTR, 0, outwh[l], DTR, 0, Lw[l][3], nullptr,
         proj, DTR, 0, nullptr, 0, 0, RPAD/16, DTR/16, ROWS, DTR, DTR, 0, 1);
    k_residual_ln<<<ROWS,256,0,stream>>>(bufA, proj, Lw[l][8], Lw[l][9], bufB, bufBh);
    // FF: 160 -> 256 (relu) -> 160
    gemm(bufBh, DTR, 0, ff1wh[l], DTR, 0, Lw[l][5], nullptr,
         nullptr, 0, 0, ffh, NHID_, 0, RPAD/16, NHID_/16, ROWS, NHID_, DTR, 1, 1);
    gemm(ffh, NHID_, 0, ff2wh[l], NHID_, 0, Lw[l][7], nullptr,
         proj, DTR, 0, nullptr, 0, 0, RPAD/16, DTR/16, ROWS, DTR, NHID_, 0, 1);
    float* f32dst = (l == 1) ? outMain : bufA;  // final layer writes `output` directly
    k_residual_ln<<<ROWS,256,0,stream>>>(bufB, proj, Lw[l][10], Lw[l][11], f32dst, bufAh);
  }

  // ---- output heads (masked by non_pad) ----
  gemm(bufAh, DTR, 0, wtypeh, DTR, 0, nullptr, nonpad,
       outType, NINP, 0, nullptr, 0, 0, RPAD/16, 3, ROWS, NINP, DTR, 0, 1);
  gemm(bufAh, DTR, 0, wtimeh, DTR, 0, nullptr, nonpad,
       outTime, 1, 0, nullptr, 0, 0, RPAD/16, 1, ROWS, 1, DTR, 0, 1);
}
